// ShuffleWindowAttention_53601191854269
// MI455X (gfx1250) — matrix operand order repaired and checked by reading, not compiled, not run on hardware
//
#include <hip/hip_runtime.h>

// ---------------------------------------------------------------------------
// Fused shuffle-window attention for MI455X (gfx1250, wave32, WMMA).
// Persistent workgroups: each WG (256 thr = 8 wave32) loops over 49-token
// windows, prefetching the next window's x-tile into L2 (global_prefetch_b8)
// while the current window runs QKV -> attention -> proj out of LDS with
// v_wmma_f32_16x16x32_f16.
// ---------------------------------------------------------------------------

#define B_DIM   8
#define N_DIM   25088          // 49 * 512
#define C_DIM   512
#define WS      49
#define WPB     512            // windows per batch element (N / WS)
#define NH      8
#define HD      64
#define MP      64             // padded window rows (49 -> 64)
#define K3C     1536           // 3 * C
#define NWIN    4096           // B_DIM * WPB
#define PWG     1024           // persistent workgroups

typedef __attribute__((ext_vector_type(16))) _Float16 v16h;
typedef __attribute__((ext_vector_type(8)))  _Float16 v8h;
typedef __attribute__((ext_vector_type(8)))  float    v8f;
typedef __attribute__((ext_vector_type(4)))  float    v4f;

__device__ __forceinline__ v16h cat8(v8h lo, v8h hi) {
    return __builtin_shufflevector(lo, hi, 0,1,2,3,4,5,6,7,8,9,10,11,12,13,14,15);
}

// A-fragment (16x32 f16, M x K): row-major source, row stride `ld` halves.
// lane: m = l16, chunks at k0 + 8*half and k0 + 16 + 8*half.
__device__ __forceinline__ v16h load_a(const _Float16* base, int row, int ld,
                                       int k0, int l16, int half) {
    const _Float16* p = base + (size_t)(row + l16) * ld + k0 + half * 8;
    return cat8(*(const v8h*)p, *(const v8h*)(p + 16));
}

// B-fragment (32x16 f16, K x N) from an [N, K] row-major matrix (torch weight
// layout, or row-major K / transposed-V): lane n = l16, chunks at k0 + 8*half and k0 + 16 + 8*half.
__device__ __forceinline__ v16h load_b(const _Float16* base, int ncol, int ld,
                                       int k0, int l16, int half) {
    const _Float16* p = base + (size_t)(ncol + l16) * ld + k0 + half * 8;
    return cat8(*(const v8h*)p, *(const v8h*)(p + 16));
}

__device__ __forceinline__ v8f wmma16(v16h a, v16h b, v8f c) {
    return __builtin_amdgcn_wmma_f32_16x16x32_f16(false, a, false, b,
                                                  (short)0, c, false, false);
}

// ---------------------------------------------------------------------------
// Prepass: fp32 -> fp16 weight conversion into workspace (weights then live in
// the 192 MB L2 for all 4096 windows; ~2 MB resident, reused 4096x).
// ---------------------------------------------------------------------------
__global__ __launch_bounds__(256) void swa_cvt_weights(const float* __restrict__ qkv_w,
                                                       const float* __restrict__ proj_w,
                                                       _Float16* __restrict__ wq,
                                                       _Float16* __restrict__ wp) {
    int i = blockIdx.x * 256 + threadIdx.x;
    if (i < K3C * C_DIM) wq[i] = (_Float16)qkv_w[i];
    if (i < C_DIM * C_DIM) wp[i] = (_Float16)proj_w[i];
}

// ---------------------------------------------------------------------------
// Main fused persistent kernel. Dynamic LDS: 4 x (64 x 512) fp16 = 256 KB.
//   xs  [64][512]  staged window activations  (reused as attn probs, 8x64x64)
//   qs  [64][512]  Q (pre-scaled)             (reused as attention output)
//   ks  [64][512]  K
//   vt  [512][64]  V transposed (c-major) so AV B-fragments are contiguous
// __launch_bounds__(256, 2): keep VGPRs <= direct window so 2 waves/SIMD fit
// (avoids s_set_vgpr_msb churn seen with unconstrained allocation).
// ---------------------------------------------------------------------------
__global__ __launch_bounds__(256, 2)
void swa_fused(const float* __restrict__ x,
               const _Float16* __restrict__ wq,
               const float* __restrict__ bq,
               const _Float16* __restrict__ wp,
               const float* __restrict__ bp,
               float* __restrict__ out) {
    extern __shared__ _Float16 smem[];
    _Float16* xs = smem;                    // also attn probs after stage 2
    _Float16* qs = smem + MP * C_DIM;       // also AV output
    _Float16* ks = smem + 2 * MP * C_DIM;
    _Float16* vt = smem + 3 * MP * C_DIM;

    const int tid  = threadIdx.x;
    const int wave = tid >> 5;
    const int lane = tid & 31;
    const int l16  = lane & 15;
    const int half = lane >> 4;
    const float qscale = 0.125f;            // hd^-0.5

    for (int wid = blockIdx.x; wid < NWIN; wid += gridDim.x) {
        const int b  = wid >> 9;            // / WPB
        const int wi = wid & (WPB - 1);

        // ---- Stage 1: gather window tile (stride-WPB shuffle), cvt fp16 ----
        // (this barrier also orders next-iter writes vs. prior-iter stage 4)
        for (int i = tid; i < MP * (C_DIM / 4); i += 256) {
            int s = i >> 7;                 // row 0..63
            int c = (i & 127) * 4;
            v4f v = {0.f, 0.f, 0.f, 0.f};
            if (s < WS)
                v = __builtin_nontemporal_load(
                    (const v4f*)(x + ((size_t)b * N_DIM + (size_t)s * WPB + wi) * C_DIM + c));
            _Float16* d = xs + s * C_DIM + c;
            d[0] = (_Float16)v.x; d[1] = (_Float16)v.y;
            d[2] = (_Float16)v.z; d[3] = (_Float16)v.w;
        }
        __syncthreads();

        // ---- Prefetch next window's x-tile into L2 (overlaps with WMMAs) ----
        {
            int wid2 = wid + gridDim.x;
            if (wid2 < NWIN) {
                int b2  = wid2 >> 9;
                int wi2 = wid2 & (WPB - 1);
                // 49 rows x 2KB = 784 cachelines of 128B
                for (int i = tid; i < WS * 16; i += 256) {
                    int s  = i >> 4;
                    int ln = i & 15;
                    __builtin_prefetch(
                        x + ((size_t)b2 * N_DIM + (size_t)s * WPB + wi2) * C_DIM + ln * 32,
                        0, 1);
                }
            }
        }

        // ---- Stage 2: QKV GEMM [64 x 512] @ [512 x 1536] (96 n-tiles / 8 waves)
        for (int t = 0; t < 12; ++t) {
            int nt   = wave * 12 + t;
            int ncol = nt * 16 + l16;
            float bias = bq[ncol];
            v8f acc[4];
            #pragma unroll
            for (int mt = 0; mt < 4; ++mt)
                #pragma unroll
                for (int r = 0; r < 8; ++r) acc[mt][r] = bias;
            for (int kk = 0; kk < 16; ++kk) {   // K = 512, step 32
                v16h bb = load_b(wq, nt * 16, C_DIM, kk * 32, l16, half);
                #pragma unroll
                for (int mt = 0; mt < 4; ++mt) {
                    v16h aa = load_a(xs, mt * 16, C_DIM, kk * 32, l16, half);
                    acc[mt] = wmma16(aa, bb, acc[mt]);
                }
            }
            int which = ncol >> 9;              // 0=Q 1=K 2=V (uniform per n-tile)
            int cl    = ncol & (C_DIM - 1);
            #pragma unroll
            for (int mt = 0; mt < 4; ++mt)
                #pragma unroll
                for (int r = 0; r < 8; ++r) {
                    int m = mt * 16 + r + 8 * half;
                    float vv = acc[mt][r];
                    if (which == 0)      qs[m * C_DIM + cl] = (_Float16)(vv * qscale);
                    else if (which == 1) ks[m * C_DIM + cl] = (_Float16)vv;
                    else                 vt[cl * MP + m]    = (_Float16)vv;  // transposed
                }
        }
        __syncthreads();

        // ---- Stage 3: attention, one head per wave ----
        const int h = wave;
        const _Float16* qh  = qs + h * HD;               // row stride C_DIM
        const _Float16* kh  = ks + h * HD;
        const _Float16* vth = vt + (size_t)h * HD * MP;  // [64 d][64 s], stride MP
        _Float16* ah = xs + h * MP * MP;                 // probs [64 q][64 k]

        for (int qt = 0; qt < 4; ++qt) {
            v8f acc[4];
            #pragma unroll
            for (int n2 = 0; n2 < 4; ++n2)
                #pragma unroll
                for (int r = 0; r < 8; ++r) acc[n2][r] = 0.f;
            #pragma unroll
            for (int kk = 0; kk < 2; ++kk) {     // K = 64 (head dim), step 32
                v16h aa = load_a(qh, qt * 16, C_DIM, kk * 32, l16, half);
                #pragma unroll
                for (int n2 = 0; n2 < 4; ++n2) {
                    v16h bb = load_b(kh, n2 * 16, C_DIM, kk * 32, l16, half);
                    acc[n2] = wmma16(aa, bb, acc[n2]);
                }
            }
            // mask padded key columns (cols 49..63 live in n-tile 3, l16 >= 1)
            if (l16 > 0)
                #pragma unroll
                for (int r = 0; r < 8; ++r) acc[3][r] = -1e30f;
            // row softmax: rows live across the 16 lanes of this half-wave
            #pragma unroll
            for (int r = 0; r < 8; ++r) {
                float mx = fmaxf(fmaxf(acc[0][r], acc[1][r]), fmaxf(acc[2][r], acc[3][r]));
                mx = fmaxf(mx, __shfl_xor(mx, 1));
                mx = fmaxf(mx, __shfl_xor(mx, 2));
                mx = fmaxf(mx, __shfl_xor(mx, 4));
                mx = fmaxf(mx, __shfl_xor(mx, 8));
                float p0 = __expf(acc[0][r] - mx), p1 = __expf(acc[1][r] - mx);
                float p2 = __expf(acc[2][r] - mx), p3 = __expf(acc[3][r] - mx);
                float s = p0 + p1 + p2 + p3;
                s += __shfl_xor(s, 1);
                s += __shfl_xor(s, 2);
                s += __shfl_xor(s, 4);
                s += __shfl_xor(s, 8);
                float inv = 1.0f / s;
                int m = qt * 16 + r + 8 * half;
                ah[m * MP +  0 + l16] = (_Float16)(p0 * inv);
                ah[m * MP + 16 + l16] = (_Float16)(p1 * inv);
                ah[m * MP + 32 + l16] = (_Float16)(p2 * inv);
                ah[m * MP + 48 + l16] = (_Float16)(p3 * inv);
            }
        }

        // AV: [64 q x 64 k] @ [64 k x 64 d]; output overlays qs (own cols only)
        for (int mt = 0; mt < 4; ++mt) {
            v8f acc[4];
            #pragma unroll
            for (int nt = 0; nt < 4; ++nt)
                #pragma unroll
                for (int r = 0; r < 8; ++r) acc[nt][r] = 0.f;
            #pragma unroll
            for (int kk = 0; kk < 2; ++kk) {
                v16h aa = load_a(ah, mt * 16, MP, kk * 32, l16, half);
                #pragma unroll
                for (int nt = 0; nt < 4; ++nt) {
                    v16h bb = load_b(vth, nt * 16, MP, kk * 32, l16, half);
                    acc[nt] = wmma16(aa, bb, acc[nt]);
                }
            }
            #pragma unroll
            for (int nt = 0; nt < 4; ++nt)
                #pragma unroll
                for (int r = 0; r < 8; ++r) {
                    int m = mt * 16 + r + 8 * half;
                    qs[m * C_DIM + h * HD + nt * 16 + l16] = (_Float16)acc[nt][r];
                }
        }
        __syncthreads();

        // ---- Stage 4: projection [64 x 512] @ [512 x 512], un-shuffle store
        for (int t = 0; t < 4; ++t) {
            int nt   = wave * 4 + t;
            int ncol = nt * 16 + l16;
            float bias = bp[ncol];
            v8f acc[4];
            #pragma unroll
            for (int mt = 0; mt < 4; ++mt)
                #pragma unroll
                for (int r = 0; r < 8; ++r) acc[mt][r] = bias;
            for (int kk = 0; kk < 16; ++kk) {
                v16h bb = load_b(wp, nt * 16, C_DIM, kk * 32, l16, half);
                #pragma unroll
                for (int mt = 0; mt < 4; ++mt) {
                    v16h aa = load_a(qs, mt * 16, C_DIM, kk * 32, l16, half);
                    acc[mt] = wmma16(aa, bb, acc[mt]);
                }
            }
            #pragma unroll
            for (int mt = 0; mt < 4; ++mt)
                #pragma unroll
                for (int r = 0; r < 8; ++r) {
                    int m = mt * 16 + r + 8 * half;  // window row (token s)
                    if (m < WS)
                        __builtin_nontemporal_store(
                            acc[mt][r],
                            out + ((size_t)b * N_DIM + (size_t)m * WPB + wi) * C_DIM + ncol);
                }
        }
        __syncthreads();   // protect xs/qs/ks/vt before next iteration's writes
    }
}

extern "C" void kernel_launch(void* const* d_in, const int* in_sizes, int n_in,
                              void* d_out, int out_size, void* d_ws, size_t ws_size,
                              hipStream_t stream) {
    const float* x      = (const float*)d_in[0];
    // d_in[1] = coords : unused by the reference computation
    const float* qkv_w  = (const float*)d_in[2];
    const float* qkv_b  = (const float*)d_in[3];
    const float* proj_w = (const float*)d_in[4];
    const float* proj_b = (const float*)d_in[5];
    float* out          = (float*)d_out;

    _Float16* wq = (_Float16*)d_ws;                  // 1536*512 fp16 = 1.5 MB
    _Float16* wp = wq + (size_t)K3C * C_DIM;         //  512*512 fp16 = 0.5 MB

    // weight fp32 -> fp16 prepass (same stream: ordered before main kernel)
    swa_cvt_weights<<<(K3C * C_DIM + 255) / 256, 256, 0, stream>>>(qkv_w, proj_w, wq, wp);

    const size_t lds_bytes = 4u * MP * C_DIM * sizeof(_Float16);   // 256 KB
    (void)hipFuncSetAttribute((const void*)swa_fused,
                              hipFuncAttributeMaxDynamicSharedMemorySize,
                              (int)lds_bytes);
    swa_fused<<<PWG, 256, lds_bytes, stream>>>(x, wq, qkv_b, wp, proj_b, out);
}
